// NoiseReduceLayer_33071248179285
// MI455X (gfx1250) — matrix operand order repaired and checked by reading, not compiled, not run on hardware
//
#include <hip/hip_runtime.h>
#include <hip/hip_bf16.h>
#include <math.h>

// ---------------------------------------------------------------------------
// Noise-reduce (non-stationary spectral gating) for MI455X / gfx1250.
// STFT and ISTFT are expressed as f16 WMMA GEMMs (v_wmma_f32_16x16x32_f16).
// B operands (framed signal / masked spectrum) are stored transposed [N][K]
// so both A and B fragments are contiguous ds_load_b128 pairs, and tiles are
// staged with gfx1250 global_load_async_to_lds_b128 + double buffering.
// ---------------------------------------------------------------------------

typedef _Float16 f16;
typedef __attribute__((ext_vector_type(16))) _Float16 v16h;
typedef __attribute__((ext_vector_type(8)))  float    v8f;

#define PI_F 3.14159265358979323846f

// Problem constants
constexpr int BATCH = 32;
constexpr int LEN   = 163840;
constexpr int NFFT  = 2048;
constexpr int HOP   = 512;
constexpr int NFREQ = 1025;          // onesided bins
constexpr int T     = 321;           // frames per batch item
constexpr int NCOL  = BATCH * T;     // 10272 total frames
constexpr int NP    = 10368;         // padded cols = 81 * 128
constexpr int IM0   = 1040;          // imag row offset (1025 -> pad to 1040)
constexpr int M1P   = 2176;          // forward GEMM rows (2080 -> 17*128)
constexpr int K2    = 2080;          // inverse GEMM K (re+im rows)

__device__ __forceinline__ float hannw(int n) {
  // periodic hann, N = 2048
  return 0.5f * (1.0f - __cosf((2.0f * PI_F / 2048.0f) * (float)n));
}

// ---------------------------------------------------------------------------
// Tiled WMMA GEMM (NT form):
//   C[M x N] (f32, row-major) = A[M x K] (f16, row-major) * Bt[N x K]^T
// block tile 128x128, BK=32, 256 threads = 8 waves in a 2x4 grid,
// each wave owns a 64x32 sub-tile = 4x2 fragments of 16x16.
// M, N multiples of 128; K a multiple of 32 (buffers padded accordingly).
// Tiles staged via async global->LDS (ASYNCcnt) with double buffering.
// ---------------------------------------------------------------------------
__global__ __launch_bounds__(256) void wmma_gemm_f16(
    const f16* __restrict__ A, const f16* __restrict__ Bt,
    float* __restrict__ C, int M, int N, int K) {
  // [buf][row][k], row stride 40 f16 = 80B (16B aligned)
  __shared__ __align__(16) f16 As[2][128][40];
  __shared__ __align__(16) f16 Bs[2][128][40];

  const int tid  = threadIdx.x;
  const int lane = tid & 31;
  const int wave = tid >> 5;
  const int hl   = lane >> 4;    // half-wave select
  const int l16  = lane & 15;
  const int wm   = wave >> 2;    // 0..1  wave row
  const int wn   = wave & 3;     // 0..3  wave col
  const int m0   = blockIdx.y * 128;
  const int n0   = blockIdx.x * 128;

  v8f acc[4][2] = {};

  // stage one 128x32 K-tile of A and Bt into LDS buffer `buf` using
  // gfx1250 async global->LDS copies (16B per lane, 4 per thread).
  auto stage = [&](int k0, int buf) {
#pragma unroll
    for (int c = 0; c < 2; ++c) {
      const int ch = tid + c * 256;       // 512 chunks of 16B per matrix
      const int r  = ch >> 2;             // tile row 0..127
      const int cc = (ch & 3) << 3;       // k offset in f16 units
      const unsigned long long ga =
          (unsigned long long)(A + (size_t)(m0 + r) * K + (k0 + cc));
      const unsigned la = (unsigned)(size_t)(&As[buf][r][cc]);
      asm volatile("global_load_async_to_lds_b128 %0, %1, off"
                   :: "v"(la), "v"(ga) : "memory");
      const unsigned long long gb =
          (unsigned long long)(Bt + (size_t)(n0 + r) * K + (k0 + cc));
      const unsigned lb = (unsigned)(size_t)(&Bs[buf][r][cc]);
      asm volatile("global_load_async_to_lds_b128 %0, %1, off"
                   :: "v"(lb), "v"(gb) : "memory");
    }
  };

  const int nk = K >> 5;
  stage(0, 0);

  for (int kt = 0; kt < nk; ++kt) {
    const int buf = kt & 1;
    if (kt + 1 < nk) {
      stage((kt + 1) << 5, buf ^ 1);
      // hint the tile after next toward the caches (global_prefetch_b8)
      if (kt + 2 < nk) {
        __builtin_prefetch(A + (size_t)(m0 + (tid >> 1)) * K + ((kt + 2) << 5), 0, 1);
        __builtin_prefetch(Bt + (size_t)(n0 + (tid >> 1)) * K + ((kt + 2) << 5), 0, 1);
      }
      // 4 newest asyncs (next tile) may stay outstanding; the current tile's
      // 4 are older and complete in order.
      asm volatile("s_wait_asynccnt 0x4" ::: "memory");
    } else {
      asm volatile("s_wait_asynccnt 0x0" ::: "memory");
    }
    __syncthreads();

    // ---- fragments per documented wave32 layouts (CDNA5 ISA)
    // A 16x32 f16: M = lane&15; elems 0..7 -> K = 8*half+e,
    //              elems 8..15 -> K = 16 + 8*half + (e-8)   (2x ds_load_b128)
    v16h af[4], bf[2];
#pragma unroll
    for (int mi = 0; mi < 4; ++mi) {
      const int row = wm * 64 + mi * 16 + l16;
#pragma unroll
      for (int e = 0; e < 8; ++e)  af[mi][e] = As[buf][row][hl * 8 + e];
#pragma unroll
      for (int e = 8; e < 16; ++e) af[mi][e] = As[buf][row][16 + hl * 8 + (e - 8)];
    }
    // B 32x16 f16: N = lane&15; elems 0..7 -> K = 8*half+e,
    //              elems 8..15 -> K = 16 + 8*half + (e-8)   (2x ds_load_b128)
#pragma unroll
    for (int ni = 0; ni < 2; ++ni) {
      const int cn = wn * 32 + ni * 16 + l16;
#pragma unroll
      for (int e = 0; e < 16; ++e) bf[ni][e] = Bs[buf][cn][(e >> 3) * 16 + hl * 8 + (e & 7)];
    }
#pragma unroll
    for (int mi = 0; mi < 4; ++mi)
#pragma unroll
      for (int ni = 0; ni < 2; ++ni)
        acc[mi][ni] = __builtin_amdgcn_wmma_f32_16x16x32_f16(
            false, af[mi], false, bf[ni], (short)0, acc[mi][ni], false, false);
    __syncthreads();   // protect buffer `buf` before it is re-staged
  }

  // ---- store: C/D 16x16 f32 layout: VGPR r -> M = r + 8*half, N = lane&15
#pragma unroll
  for (int mi = 0; mi < 4; ++mi)
#pragma unroll
    for (int ni = 0; ni < 2; ++ni) {
      const int rb = m0 + wm * 64 + mi * 16 + hl * 8;
      const int cn = n0 + wn * 32 + ni * 16 + l16;
#pragma unroll
      for (int r = 0; r < 8; ++r)
        C[(size_t)(rb + r) * N + cn] = acc[mi][ni][r];
    }
}

// ---------------------------------------------------------------------------
// Basis builders (unnormalized DFT; scale cancels through the mask, 1/N is
// folded into the overlap-add).
// ---------------------------------------------------------------------------
__global__ void k_build_W(f16* __restrict__ W) {
  const size_t total = (size_t)M1P * NFFT;
  const size_t stride = (size_t)gridDim.x * blockDim.x;
  for (size_t i = (size_t)blockIdx.x * blockDim.x + threadIdx.x; i < total; i += stride) {
    const int m = (int)(i >> 11);     // / 2048
    const int k = (int)(i & 2047);
    float v = 0.0f;
    if (m < NFREQ) {                              // real rows: cos * hann
      const int ph = (m * k) & 2047;
      v = __cosf((2.0f * PI_F / 2048.0f) * (float)ph) * hannw(k);
    } else if (m >= IM0 && m < IM0 + NFREQ) {     // imag rows: -sin * hann
      const int f = m - IM0;
      const int ph = (f * k) & 2047;
      v = -__sinf((2.0f * PI_F / 2048.0f) * (float)ph) * hannw(k);
    }
    W[i] = (f16)v;
  }
}

__global__ void k_build_V(f16* __restrict__ V) {
  const size_t total = (size_t)NFFT * K2;
  const size_t stride = (size_t)gridDim.x * blockDim.x;
  for (size_t i = (size_t)blockIdx.x * blockDim.x + threadIdx.x; i < total; i += stride) {
    const int fr = (int)(i % K2);
    const int n  = (int)(i / K2);
    float v = 0.0f;
    if (fr < NFREQ) {
      const float c = (fr == 0 || fr == 1024) ? 1.0f : 2.0f;
      const int ph = (n * fr) & 2047;
      v = c * __cosf((2.0f * PI_F / 2048.0f) * (float)ph) * hannw(n);
    } else if (fr >= IM0 && fr < IM0 + NFREQ) {
      const int f = fr - IM0;
      const float c = (f == 0 || f == 1024) ? 1.0f : 2.0f;
      const int ph = (n * f) & 2047;
      v = -c * __sinf((2.0f * PI_F / 2048.0f) * (float)ph) * hannw(n);
    }
    V[i] = (f16)v;
  }
}

// Frame the padded signal into Xt[NP][2048] (transposed; boundary='zeros'
// pads 1024 each side). Contiguous reads of x, contiguous f16 writes.
__global__ void k_frame(const float* __restrict__ x, f16* __restrict__ Xt) {
  const size_t total = (size_t)NP * NFFT;
  const size_t stride = (size_t)gridDim.x * blockDim.x;
  for (size_t i = (size_t)blockIdx.x * blockDim.x + threadIdx.x; i < total; i += stride) {
    const int k   = (int)(i & 2047);
    const int col = (int)(i >> 11);
    float v = 0.0f;
    if (col < NCOL) {
      const int b = col / T, t = col % T;
      const int s = t * HOP + k - 1024;
      if (s >= 0 && s < LEN) v = x[(size_t)b * LEN + s];
    }
    Xt[i] = (f16)v;
  }
}

// |S| from stacked re/im GEMM output
__global__ void k_mag(const float* __restrict__ S, float* __restrict__ ab) {
  const size_t total = (size_t)BATCH * NFREQ * T;
  const size_t stride = (size_t)gridDim.x * blockDim.x;
  for (size_t i = (size_t)blockIdx.x * blockDim.x + threadIdx.x; i < total; i += stride) {
    const int t = (int)(i % T);
    const size_t bf = i / T;
    const int f = (int)(bf % NFREQ);
    const int b = (int)(bf / NFREQ);
    const size_t col = (size_t)b * T + t;
    const float re = S[(size_t)f * NP + col];
    const float im = S[(size_t)(IM0 + f) * NP + col];
    ab[i] = sqrtf(re * re + im * im);
  }
}

// forward+backward EMA (filtfilt, y[-1]=x[0] init) -> sigmoid gate
__global__ void k_ema_mask(const float* __restrict__ ab, float* __restrict__ tb,
                           float* __restrict__ mb) {
  const int row = blockIdx.x * blockDim.x + threadIdx.x;
  if (row >= BATCH * NFREQ) return;
  const size_t base = (size_t)row * T;
  const float tf = 2.0f * 16000.0f / 512.0f;                        // 62.5 frames
  const float be = (sqrtf(1.0f + 4.0f * tf * tf) - 1.0f) / (2.0f * tf * tf);
  // forward pass
  float y = ab[base];
  tb[base] = y;
  for (int t = 1; t < T; ++t) {
    y = be * ab[base + t] + (1.0f - be) * y;
    tb[base + t] = y;
  }
  // backward pass + gate
  y = tb[base + T - 1];
  for (int t = T - 1; t >= 0; --t) {
    y = be * tb[base + t] + (1.0f - be) * y;
    const float sm = y;
    const float a  = ab[base + t];
    const float sig = (a - sm) / sm;                 // sig_mult_above
    mb[base + t] = 1.0f / (1.0f + __expf(-(sig - 2.0f) * 10.0f));
  }
}

// separable smoothing: 65-tap triangle along freq ...
__global__ void k_conv_f(const float* __restrict__ m, float* __restrict__ o) {
  const size_t total = (size_t)BATCH * NFREQ * T;
  const size_t stride = (size_t)gridDim.x * blockDim.x;
  for (size_t i = (size_t)blockIdx.x * blockDim.x + threadIdx.x; i < total; i += stride) {
    const int t = (int)(i % T);
    const size_t bf = i / T;
    const int f = (int)(bf % NFREQ);
    const int b = (int)(bf / NFREQ);
    float acc = 0.0f;
    for (int j = -32; j <= 32; ++j) {
      const int fj = f + j;
      if (fj < 0 || fj >= NFREQ) continue;
      const int aj = j < 0 ? -j : j;
      acc += (float)(33 - aj) * (1.0f / 33.0f) * m[((size_t)b * NFREQ + fj) * T + t];
    }
    o[i] = acc;
  }
}

// ... then {0.5,1,0.5} along time, with the filter-sum (66) normalization
__global__ void k_conv_t(const float* __restrict__ m, float* __restrict__ o) {
  const size_t total = (size_t)BATCH * NFREQ * T;
  const size_t stride = (size_t)gridDim.x * blockDim.x;
  for (size_t i = (size_t)blockIdx.x * blockDim.x + threadIdx.x; i < total; i += stride) {
    const int t = (int)(i % T);
    const size_t bf = i / T;
    float acc = 0.0f;
    if (t - 1 >= 0) acc += 0.5f * m[bf * T + t - 1];
    acc += m[bf * T + t];
    if (t + 1 < T)  acc += 0.5f * m[bf * T + t + 1];
    o[i] = acc * (1.0f / 66.0f);
  }
}

// masked spectrum -> transposed f16 operand Yt[NP][K2] for the inverse GEMM
__global__ void k_build_Y(const float* __restrict__ S, const float* __restrict__ mk,
                          f16* __restrict__ Yt) {
  const size_t total = (size_t)NP * K2;
  const size_t stride = (size_t)gridDim.x * blockDim.x;
  for (size_t i = (size_t)blockIdx.x * blockDim.x + threadIdx.x; i < total; i += stride) {
    const int r   = (int)(i % K2);
    const int col = (int)(i / K2);
    float v = 0.0f;
    if (col < NCOL) {
      int f = -1;
      if (r < NFREQ) f = r;
      else if (r >= IM0 && r < IM0 + NFREQ) f = r - IM0;
      if (f >= 0) {
        const int b = col / T, t = col % T;
        const float s = S[(size_t)r * NP + col];
        v = s * mk[((size_t)b * NFREQ + f) * T + t];
      }
    }
    Yt[i] = (f16)v;
  }
}

// overlap-add with hann^2 normalization; 1/NFFT irfft factor folded in;
// boundary trim of 1024 and crop to LEN are implicit in the index mapping.
__global__ void k_ola(const float* __restrict__ Z, float* __restrict__ out) {
  const size_t total = (size_t)BATCH * LEN;
  const size_t stride = (size_t)gridDim.x * blockDim.x;
  for (size_t i = (size_t)blockIdx.x * blockDim.x + threadIdx.x; i < total; i += stride) {
    const int s = (int)(i % LEN);
    const int b = (int)(i / LEN);
    const int p = s + 1024;
    const int thi = p >> 9;            // p / HOP
    float num = 0.0f, den = 0.0f;
    for (int t = thi - 3; t <= thi; ++t) {
      if (t < 0 || t >= T) continue;
      const int n = p - (t << 9);
      if (n < 0 || n >= NFFT) continue;
      const float w = hannw(n);
      num += Z[(size_t)n * NP + (size_t)b * T + t];   // window already in V
      den += w * w;
    }
    out[i] = num / (2048.0f * den);
  }
}

// ---------------------------------------------------------------------------
extern "C" void kernel_launch(void* const* d_in, const int* in_sizes, int n_in,
                              void* d_out, int out_size, void* d_ws, size_t ws_size,
                              hipStream_t stream) {
  const float* x = (const float*)d_in[0];
  float* out = (float*)d_out;
  (void)in_sizes; (void)n_in; (void)out_size; (void)ws_size;

  // workspace carve (256B aligned); X region is reused for Y, S region for Z
  char* p = (char*)d_ws;
  auto up = [](size_t v) { return (v + 255) & ~(size_t)255; };
  f16*   W  = (f16*)p;   p += up((size_t)M1P * NFFT * sizeof(f16));
  f16*   V  = (f16*)p;   p += up((size_t)NFFT * K2 * sizeof(f16));
  f16*   XY = (f16*)p;   p += up((size_t)NP * K2 * sizeof(f16));       // max(Xt, Yt)
  float* SZ = (float*)p; p += up((size_t)M1P * NP * sizeof(float));    // max(S, Z)
  float* ab = (float*)p; p += up((size_t)BATCH * NFREQ * T * sizeof(float));
  float* tb = (float*)p; p += up((size_t)BATCH * NFREQ * T * sizeof(float));
  float* mb = (float*)p; p += up((size_t)BATCH * NFREQ * T * sizeof(float));

  const dim3 blk(256);
  k_build_W<<<4096, blk, 0, stream>>>(W);
  k_build_V<<<4096, blk, 0, stream>>>(V);
  k_frame<<<8192, blk, 0, stream>>>(x, XY);

  // forward STFT:  S[2176 x 10368] = W[2176 x 2048] * Xt[10368 x 2048]^T
  wmma_gemm_f16<<<dim3(NP / 128, M1P / 128), blk, 0, stream>>>(W, XY, SZ, M1P, NP, NFFT);

  k_mag<<<8192, blk, 0, stream>>>(SZ, ab);
  k_ema_mask<<<(BATCH * NFREQ + 255) / 256, blk, 0, stream>>>(ab, tb, mb);
  k_conv_f<<<8192, blk, 0, stream>>>(mb, tb);
  k_conv_t<<<8192, blk, 0, stream>>>(tb, ab);            // final mask -> ab
  k_build_Y<<<8192, blk, 0, stream>>>(SZ, ab, XY);       // Yt overlays Xt region

  // inverse STFT:  Z[2048 x 10368] = V[2048 x 2080] * Yt[10368 x 2080]^T
  wmma_gemm_f16<<<dim3(NP / 128, NFFT / 128), blk, 0, stream>>>(V, XY, SZ, NFFT, NP, K2);

  k_ola<<<(BATCH * LEN + 255) / 256, blk, 0, stream>>>(SZ, out);
}
